// DGCNN_4037269258632
// MI455X (gfx1250) — matrix operand order repaired and checked by reading, not compiled, not run on hardware
//
#include <hip/hip_runtime.h>

#define KNNK 10
#define SLOPE 0.2f

typedef __attribute__((ext_vector_type(16))) __bf16 v16bf;
typedef __attribute__((ext_vector_type(16))) unsigned short v16us;
typedef __attribute__((ext_vector_type(8))) unsigned short v8us;
typedef __attribute__((ext_vector_type(8))) float v8f;

union BfVec { v16us u; v16bf h; };

__device__ __forceinline__ unsigned short f2bf(float f) {
  unsigned int u = __float_as_uint(f);
  u += 0x7FFFu + ((u >> 16) & 1u);   // round-to-nearest-even
  return (unsigned short)(u >> 16);
}

__device__ __forceinline__ float bf2f(unsigned short u) {
  return __uint_as_float((unsigned int)u << 16);
}

__device__ __forceinline__ v8f wmma_bf16(v16us a, v16us b, v8f c) {
  BfVec av, bv;
  av.u = a; bv.u = b;
  return __builtin_amdgcn_wmma_f32_16x16x32_bf16(
      false, av.h, false, bv.h, (short)0, c, false, false);
}

// A-fragment: two 8-element runs at p[0..7] and p[16..23]  (16B-aligned)
__device__ __forceinline__ v16us loadA16(const unsigned short* __restrict__ p) {
  v8us lo = *(const v8us*)(p);
  v8us hi = *(const v8us*)(p + 16);
  v16us r;
#pragma unroll
  for (int e = 0; e < 8; ++e) { r[e] = lo[e]; r[8 + e] = hi[e]; }
  return r;
}

// B-fragment: two 8-element runs at p[0..7] and p[16..23]  (16B-aligned)
__device__ __forceinline__ v16us loadB16(const unsigned short* __restrict__ p) {
  v8us lo = *(const v8us*)(p);
  v8us hi = *(const v8us*)(p + 16);
  v16us r;
#pragma unroll
  for (int e = 0; e < 8; ++e) { r[e] = lo[e]; r[8 + e] = hi[e]; }
  return r;
}

// ---------------- prep kernels ----------------

// x (B,3,N) -> xb bf16 padded (B,N,32), sq (B,N)
__global__ void prep1_kernel(const float* __restrict__ x,
                             unsigned short* __restrict__ xb, float* __restrict__ sq,
                             int B, int N) {
  int i = blockIdx.x * blockDim.x + threadIdx.x;
  if (i >= B * N) return;
  int b = i / N, n = i - b * N;
  const float* xp = x + (size_t)b * 3 * N + n;
  float v0 = xp[0], v1 = xp[N], v2 = xp[2 * N];
  unsigned short* p = xb + (size_t)i * 32;
  p[0] = f2bf(v0); p[1] = f2bf(v1); p[2] = f2bf(v2);
  for (int c = 3; c < 32; ++c) p[c] = 0;
  sq[i] = v0 * v0 + v1 * v1 + v2 * v2;
}

// squared norms of a bf16 feature region (row stride fstride, column offset foff)
__global__ void sqn_kernel(const unsigned short* __restrict__ fb, int fstride, int foff,
                           int C, float* __restrict__ sq, int BN) {
  int i = blockIdx.x * blockDim.x + threadIdx.x;
  if (i >= BN) return;
  const unsigned short* p = fb + (size_t)i * fstride + foff;
  float s = 0.f;
  for (int c = 0; c < C; ++c) { float v = bf2f(p[c]); s += v * v; }
  sq[i] = s;
}

// f32 -> bf16 weight conversion (run once per weight tensor)
__global__ void convw_kernel(const float* __restrict__ src,
                             unsigned short* __restrict__ dst, int nelem) {
  int i = blockIdx.x * blockDim.x + threadIdx.x;
  if (i < nelem) dst[i] = f2bf(src[i]);
}

// ---------------- KNN via bf16 WMMA Gram tiles ----------------
// grid: B*(N/128) blocks of 256 threads (8 waves); each wave owns 16 rows.
template <int NCHUNK>
__global__ void knn_kernel(const unsigned short* __restrict__ fb, int fstride, int foff,
                           const float* __restrict__ sq,
                           int* __restrict__ knnidx, int N) {
  __shared__ float tile[8][16][16];
  int lane = threadIdx.x & 31;
  int wave = threadIdx.x >> 5;
  int blocksPerBatch = N >> 7;
  int b = blockIdx.x / blocksPerBatch;
  int m0 = (blockIdx.x - b * blocksPerBatch) * 128 + wave * 16;
  const float* SQ = sq + (size_t)b * N;
  int l15 = lane & 15;
  int half = lane >> 4;
  int ml = m0 + l15;
  float sqm = SQ[ml];
  int abase = half * 8;
  int kb = half * 8;
  const unsigned short* Xm = fb + ((size_t)b * N + ml) * fstride + foff;
  v16us afrag[NCHUNK];
#pragma unroll
  for (int ck = 0; ck < NCHUNK; ++ck)
    afrag[ck] = loadA16(Xm + ck * 32 + abase);

  float bd[KNNK]; int bi[KNNK];
#pragma unroll
  for (int t = 0; t < KNNK; ++t) { bd[t] = -3.0e38f; bi[t] = 0; }

  for (int n0 = 0; n0 < N; n0 += 16) {
    int bn = n0 + l15;
    const unsigned short* Xn = fb + ((size_t)b * N + bn) * fstride + foff;
    v8f acc;
#pragma unroll
    for (int e = 0; e < 8; ++e) acc[e] = 0.f;
#pragma unroll
    for (int ck = 0; ck < NCHUNK; ++ck)
      acc = wmma_bf16(afrag[ck], loadB16(Xn + ck * 32 + kb), acc);

    float sqn = SQ[bn];
#pragma unroll
    for (int j = 0; j < 8; ++j) {
      int mrow = j + half * 8;
      float sqr = __shfl(sqm, mrow, 32);
      tile[wave][mrow][l15] = 2.0f * acc[j] - sqr - sqn;
    }
    __syncthreads();
    if (lane < 16) {
      for (int t = 0; t < 16; ++t) {
        float v = tile[wave][lane][t];
        if (v > bd[KNNK - 1]) {
          bd[KNNK - 1] = v; bi[KNNK - 1] = n0 + t;
#pragma unroll
          for (int s = KNNK - 1; s > 0; --s) {
            if (bd[s] > bd[s - 1]) {
              float tv = bd[s]; bd[s] = bd[s - 1]; bd[s - 1] = tv;
              int ti = bi[s]; bi[s] = bi[s - 1]; bi[s - 1] = ti;
            }
          }
        }
      }
    }
    __syncthreads();
  }
  if (lane < 16) {
    int* op = knnidx + ((size_t)b * N + ml) * KNNK;
#pragma unroll
    for (int t = 0; t < KNNK; ++t) op[t] = bi[t];
  }
}

// ---------------- EdgeConv: gather + GEMM + BN/LReLU + max over k ----------------
// One wave per (b, n, group-of-64 outputs). M-tile rows = 10 neighbors of n
// (rows 10..15 duplicate kk=0..5; harmless for max). Max over k: in-register +
// shfl_xor(16) over the D-tile M dimension. Output written as bf16 into the
// concatenated point_feat buffer (stride 512) at column offset pfoff.
template <int CIN>
__global__ void edgeconv_kernel(const unsigned short* __restrict__ fb, int fstride, int foff,
                                const int* __restrict__ knnidx,
                                const unsigned short* __restrict__ wb,
                                const float* __restrict__ gg,
                                const float* __restrict__ bb,
                                unsigned short* __restrict__ pfb, int pfoff,
                                int N, int O) {
  constexpr int K2 = 2 * CIN;
  constexpr int NCH = (K2 + 31) / 32;
  const int NT = 4;
  int gw = blockIdx.x * (blockDim.x >> 5) + (threadIdx.x >> 5);
  int lane = threadIdx.x & 31;
  int ng = O >> 6;
  int b = gw / (N * ng);
  int rem = gw - b * (N * ng);
  int n = rem / ng;
  int og = rem - n * ng;
  int o0base = og * 64;
  int l15 = lane & 15;
  int half = lane >> 4;
  int kk = l15 % KNNK;
  int nbr = knnidx[((size_t)b * N + n) * KNNK + kk];
  const unsigned short* frN = fb + ((size_t)b * N + nbr) * fstride + foff;
  const unsigned short* frC = fb + ((size_t)b * N + n) * fstride + foff;
  int abase = half * 8;
  int kbB = half * 8;
  v8f acc[NT];
#pragma unroll
  for (int t = 0; t < NT; ++t)
#pragma unroll
    for (int e = 0; e < 8; ++e) acc[t][e] = 0.f;
#pragma unroll
  for (int ck = 0; ck < NCH; ++ck) {
    v16us a;
    if (K2 % 32 == 0) {          // chunk wholly inside one region (layers 2-4)
      int cbase = ck * 32;
      if (cbase < CIN) {         // neighbor - center region
        const unsigned short* pn = frN + cbase + abase;
        const unsigned short* pc = frC + cbase + abase;
        v8us n0 = *(const v8us*)pn, n1 = *(const v8us*)(pn + 16);
        v8us c0 = *(const v8us*)pc, c1 = *(const v8us*)(pc + 16);
#pragma unroll
        for (int e = 0; e < 8; ++e) {
          a[e]     = f2bf(bf2f(n0[e]) - bf2f(c0[e]));
          a[8 + e] = f2bf(bf2f(n1[e]) - bf2f(c1[e]));
        }
      } else {                   // center region: raw bf16 copy
        a = loadA16(frC + (cbase - CIN) + abase);
      }
    } else {                     // generic guarded path (layer 1, K2=6)
#pragma unroll
      for (int e = 0; e < 16; ++e) {
        int c = ck * 32 + abase + (e < 8 ? e : e + 8);
        float v = 0.f;
        if (c < CIN) v = bf2f(frN[c]) - bf2f(frC[c]);
        else if (c < K2) v = bf2f(frC[c - CIN]);
        a[e] = f2bf(v);
      }
    }
#pragma unroll
    for (int t = 0; t < NT; ++t) {
      int o = o0base + t * 16 + l15;
      v16us bm;
      if (K2 % 32 == 0) {
        bm = loadB16(wb + (size_t)o * K2 + ck * 32 + kbB);
      } else {
#pragma unroll
        for (int e = 0; e < 16; ++e) {
          int c = ck * 32 + kbB + (e < 8 ? e : e + 8);
          bm[e] = (c < K2) ? wb[(size_t)o * K2 + c] : (unsigned short)0;
        }
      }
      acc[t] = wmma_bf16(a, bm, acc[t]);
    }
  }
  float rsq = rsqrtf(1.0f + 1e-5f);
#pragma unroll
  for (int t = 0; t < NT; ++t) {
    int o = o0base + t * 16 + l15;
    float s = gg[o] * rsq;
    float bias = bb[o];
    float m = -3.0e38f;
#pragma unroll
    for (int j = 0; j < 8; ++j) {
      float y = acc[t][j] * s + bias;
      y = y > 0.f ? y : SLOPE * y;
      m = fmaxf(m, y);
    }
    m = fmaxf(m, __shfl_xor(m, 16, 32));
    if (lane < 16) pfb[((size_t)b * N + n) * 512 + pfoff + o] = f2bf(m);
  }
}

// ---------------- conv5: point_feat(512) -> h(128) with BN/LReLU ----------------

__global__ void conv5_kernel(const unsigned short* __restrict__ pfb,
                             const unsigned short* __restrict__ wb,
                             const float* __restrict__ gg, const float* __restrict__ bb,
                             float* __restrict__ h) {
  const int K = 512, O = 128, NT = 4;
  int gw = blockIdx.x * (blockDim.x >> 5) + (threadIdx.x >> 5);
  int lane = threadIdx.x & 31;
  int rt = gw >> 1, og = gw & 1;
  int r0 = rt * 16, o0base = og * 64;
  int l15 = lane & 15, half = lane >> 4;
  size_t rl = (size_t)(r0 + l15);
  int abase = half * 8, kbB = half * 8;
  const unsigned short* arow = pfb + rl * 512;
  v8f acc[NT];
#pragma unroll
  for (int t = 0; t < NT; ++t)
#pragma unroll
    for (int e = 0; e < 8; ++e) acc[t][e] = 0.f;
#pragma unroll
  for (int ck = 0; ck < K / 32; ++ck) {
    v16us a = loadA16(arow + ck * 32 + abase);
#pragma unroll
    for (int t = 0; t < NT; ++t) {
      int o = o0base + t * 16 + l15;
      acc[t] = wmma_bf16(a, loadB16(wb + (size_t)o * K + ck * 32 + kbB), acc[t]);
    }
  }
  float rsq = rsqrtf(1.0f + 1e-5f);
#pragma unroll
  for (int t = 0; t < NT; ++t) {
    int o = o0base + t * 16 + l15;
    float s = gg[o] * rsq;
    float bias = bb[o];
#pragma unroll
    for (int j = 0; j < 8; ++j) {
      int row = r0 + j + half * 8;
      float y = acc[t][j] * s + bias;
      y = y > 0.f ? y : SLOPE * y;
      h[(size_t)row * O + o] = y;
    }
  }
}

// ---------------- final: concat(point_feat, g) (768) -> out (128) ----------------

__global__ void final_kernel(const unsigned short* __restrict__ pfb,
                             const unsigned short* __restrict__ gfb,
                             const unsigned short* __restrict__ wb,
                             float* __restrict__ out, int N) {
  const int K = 768, O = 128, NT = 4;
  int gw = blockIdx.x * (blockDim.x >> 5) + (threadIdx.x >> 5);
  int lane = threadIdx.x & 31;
  int rt = gw >> 1, og = gw & 1;
  int r0 = rt * 16, o0base = og * 64;
  int l15 = lane & 15, half = lane >> 4;
  size_t rl = (size_t)(r0 + l15);
  int abase = half * 8, kbB = half * 8;
  const unsigned short* arow = pfb + rl * 512;
  const unsigned short* grow = gfb + (size_t)(rl / (size_t)N) * 256;
  v8f acc[NT];
#pragma unroll
  for (int t = 0; t < NT; ++t)
#pragma unroll
    for (int e = 0; e < 8; ++e) acc[t][e] = 0.f;
#pragma unroll
  for (int ck = 0; ck < K / 32; ++ck) {
    v16us a = (ck * 32 < 512) ? loadA16(arow + ck * 32 + abase)
                              : loadA16(grow + (ck * 32 - 512) + abase);
#pragma unroll
    for (int t = 0; t < NT; ++t) {
      int o = o0base + t * 16 + l15;
      acc[t] = wmma_bf16(a, loadB16(wb + (size_t)o * K + ck * 32 + kbB), acc[t]);
    }
  }
#pragma unroll
  for (int t = 0; t < NT; ++t) {
    int o = o0base + t * 16 + l15;
#pragma unroll
    for (int j = 0; j < 8; ++j) {
      int row = r0 + j + half * 8;
      out[(size_t)row * O + o] = acc[t][j];
    }
  }
}

// ---------------- pooling + FC ----------------

__global__ void pool_kernel(const float* __restrict__ h, float* __restrict__ g, int N) {
  __shared__ float smax[256];
  __shared__ float ssum[256];
  int b = blockIdx.x >> 7;
  int o = blockIdx.x & 127;
  int tid = threadIdx.x;
  float mx = -3.0e38f, sm = 0.f;
  for (int n = tid; n < N; n += 256) {
    float v = h[((size_t)b * N + n) * 128 + o];
    mx = fmaxf(mx, v);
    sm += v;
  }
  smax[tid] = mx; ssum[tid] = sm;
  __syncthreads();
  for (int s = 128; s > 0; s >>= 1) {
    if (tid < s) { smax[tid] = fmaxf(smax[tid], smax[tid + s]); ssum[tid] += ssum[tid + s]; }
    __syncthreads();
  }
  if (tid == 0) {
    g[b * 256 + o] = smax[0];
    g[b * 256 + 128 + o] = ssum[0] / (float)N;
  }
}

__global__ void fc1_kernel(const float* __restrict__ g, const float* __restrict__ wl1,
                           const float* __restrict__ g6, const float* __restrict__ b6,
                           float* __restrict__ o1) {
  int i = blockIdx.x * blockDim.x + threadIdx.x;
  if (i >= 8 * 512) return;
  int b = i >> 9, o = i & 511;
  float s = 0.f;
  for (int c = 0; c < 256; ++c) s += g[b * 256 + c] * wl1[o * 256 + c];
  float y = s * (g6[o] * rsqrtf(1.0f + 1e-5f)) + b6[o];
  o1[i] = y > 0.f ? y : SLOPE * y;
}

// second FC writes bf16 directly (consumed only by final_kernel A-fragments)
__global__ void fc2_kernel(const float* __restrict__ o1, const float* __restrict__ wl2,
                           const float* __restrict__ bl2, const float* __restrict__ g7,
                           const float* __restrict__ b7, unsigned short* __restrict__ gfb) {
  int i = blockIdx.x * blockDim.x + threadIdx.x;
  if (i >= 8 * 256) return;
  int b = i >> 8, o = i & 255;
  float s = 0.f;
  for (int c = 0; c < 512; ++c) s += o1[b * 512 + c] * wl2[o * 512 + c];
  s += bl2[o];
  float y = s * (g7[o] * rsqrtf(1.0f + 1e-5f)) + b7[o];
  gfb[i] = f2bf(y > 0.f ? y : SLOPE * y);
}

// ---------------- launch ----------------

extern "C" void kernel_launch(void* const* d_in, const int* in_sizes, int n_in,
                              void* d_out, int out_size, void* d_ws, size_t ws_size,
                              hipStream_t stream) {
  (void)in_sizes; (void)n_in; (void)out_size; (void)ws_size;
  const int B = 8, N = 2048, BN = B * N;
  const float* x   = (const float*)d_in[0];
  const float* w1  = (const float*)d_in[1];
  const float* g1  = (const float*)d_in[2];
  const float* b1  = (const float*)d_in[3];
  const float* w2  = (const float*)d_in[4];
  const float* g2  = (const float*)d_in[5];
  const float* b2  = (const float*)d_in[6];
  const float* w3  = (const float*)d_in[7];
  const float* g3  = (const float*)d_in[8];
  const float* b3  = (const float*)d_in[9];
  const float* w4  = (const float*)d_in[10];
  const float* g4  = (const float*)d_in[11];
  const float* b4  = (const float*)d_in[12];
  const float* w5  = (const float*)d_in[13];
  const float* g5  = (const float*)d_in[14];
  const float* b5  = (const float*)d_in[15];
  const float* wl1 = (const float*)d_in[16];
  const float* g6  = (const float*)d_in[17];
  const float* b6  = (const float*)d_in[18];
  const float* wl2 = (const float*)d_in[19];
  const float* bl2 = (const float*)d_in[20];
  const float* g7  = (const float*)d_in[21];
  const float* b7  = (const float*)d_in[22];
  const float* wl3 = (const float*)d_in[23];

  char* p = (char*)d_ws;
  auto walloc = [&](size_t bytes) {
    void* r = (void*)p;
    p += (bytes + 255) & ~(size_t)255;
    return r;
  };
  unsigned short* xb   = (unsigned short*)walloc((size_t)BN * 32 * 2);
  unsigned short* pfb  = (unsigned short*)walloc((size_t)BN * 512 * 2);
  float*          sq   = (float*)walloc((size_t)BN * 4);
  int*            idx  = (int*)walloc((size_t)BN * KNNK * 4);
  float*          hb   = (float*)walloc((size_t)BN * 128 * 4);
  float*          gp   = (float*)walloc(8 * 256 * 4);
  float*          o1b  = (float*)walloc(8 * 512 * 4);
  unsigned short* gfb  = (unsigned short*)walloc(8 * 256 * 2);
  unsigned short* w1b  = (unsigned short*)walloc(64 * 6 * 2);
  unsigned short* w2b  = (unsigned short*)walloc(64 * 128 * 2);
  unsigned short* w3b  = (unsigned short*)walloc(128 * 128 * 2);
  unsigned short* w4b  = (unsigned short*)walloc(256 * 256 * 2);
  unsigned short* w5b  = (unsigned short*)walloc(128 * 512 * 2);
  unsigned short* wl3b = (unsigned short*)walloc(128 * 768 * 2);

  int prepBlocks = (BN + 255) / 256;
  int knnBlocks = B * (N / 128);

  // one-time weight conversions to bf16
  convw_kernel<<<(64 * 6 + 255) / 256, 256, 0, stream>>>(w1, w1b, 64 * 6);
  convw_kernel<<<(64 * 128 + 255) / 256, 256, 0, stream>>>(w2, w2b, 64 * 128);
  convw_kernel<<<(128 * 128 + 255) / 256, 256, 0, stream>>>(w3, w3b, 128 * 128);
  convw_kernel<<<(256 * 256 + 255) / 256, 256, 0, stream>>>(w4, w4b, 256 * 256);
  convw_kernel<<<(128 * 512 + 255) / 256, 256, 0, stream>>>(w5, w5b, 128 * 512);
  convw_kernel<<<(128 * 768 + 255) / 256, 256, 0, stream>>>(wl3, wl3b, 128 * 768);

  // Layer 1: C=3 (padded to 32) -> 64   (point_feat cols [0,64))
  prep1_kernel<<<prepBlocks, 256, 0, stream>>>(x, xb, sq, B, N);
  knn_kernel<1><<<knnBlocks, 256, 0, stream>>>(xb, 32, 0, sq, idx, N);
  edgeconv_kernel<3><<<BN * 1 / 4, 128, 0, stream>>>(xb, 32, 0, idx, w1b, g1, b1,
                                                     pfb, 0, N, 64);

  // Layer 2: C=64 -> 64   (cols [64,128))
  sqn_kernel<<<prepBlocks, 256, 0, stream>>>(pfb, 512, 0, 64, sq, BN);
  knn_kernel<2><<<knnBlocks, 256, 0, stream>>>(pfb, 512, 0, sq, idx, N);
  edgeconv_kernel<64><<<BN * 1 / 4, 128, 0, stream>>>(pfb, 512, 0, idx, w2b, g2, b2,
                                                      pfb, 64, N, 64);

  // Layer 3: C=64 -> 128  (cols [128,256))
  sqn_kernel<<<prepBlocks, 256, 0, stream>>>(pfb, 512, 64, 64, sq, BN);
  knn_kernel<2><<<knnBlocks, 256, 0, stream>>>(pfb, 512, 64, sq, idx, N);
  edgeconv_kernel<64><<<BN * 2 / 4, 128, 0, stream>>>(pfb, 512, 64, idx, w3b, g3, b3,
                                                      pfb, 128, N, 128);

  // Layer 4: C=128 -> 256 (cols [256,512))
  sqn_kernel<<<prepBlocks, 256, 0, stream>>>(pfb, 512, 128, 128, sq, BN);
  knn_kernel<4><<<knnBlocks, 256, 0, stream>>>(pfb, 512, 128, sq, idx, N);
  edgeconv_kernel<128><<<BN * 4 / 4, 128, 0, stream>>>(pfb, 512, 128, idx, w4b, g4, b4,
                                                       pfb, 256, N, 256);

  // conv5: point_feat(512) -> h(128)
  conv5_kernel<<<(BN / 16) * 2 / 4, 128, 0, stream>>>(pfb, w5b, g5, b5, hb);

  // global max+avg pooling -> (B,256)
  pool_kernel<<<8 * 128, 256, 0, stream>>>(hb, gp, N);

  // FC layers (tiny, scalar)
  fc1_kernel<<<(8 * 512 + 127) / 128, 128, 0, stream>>>(gp, wl1, g6, b6, o1b);
  fc2_kernel<<<(8 * 256 + 127) / 128, 128, 0, stream>>>(o1b, wl2, bl2, g7, b7, gfb);

  // final projection: concat(point_feat, g) (768) -> out(128)
  final_kernel<<<(BN / 16) * 2 / 4, 128, 0, stream>>>(pfb, gfb, wl3b,
                                                      (float*)d_out, N);
}
